// ASS_JRG_3573412790879
// MI455X (gfx1250) — matrix operand order repaired and checked by reading, not compiled, not run on hardware
//
#include <hip/hip_runtime.h>

typedef _Float16 v16h __attribute__((ext_vector_type(16)));
typedef _Float16 v8h  __attribute__((ext_vector_type(8)));
typedef float    v8f  __attribute__((ext_vector_type(8)));

#define B_    16
#define J_    17
#define T_    12
#define D_    400
#define H_    4
#define HALF_ 200
#define H1_   128
#define KP_   224      // K padded to 7*32 per idx slice
#define RS_   448      // f16 row stride = 2 * KP_ (idx0 slice | idx1 slice)

// padded row counts (multiples of 128 = rows per GEMM block)
#define RW_PAD_   256     // 192  -> 256
#define RC0_PAD_  3328    // 3264 -> 3328
#define RBIG_PAD_ 13056   // already multiple of 128

// ---------------- zero-fill the whole f16 intermediate region ----------------
__global__ void k_zero(uint4* __restrict__ p, int n) {
  int i = blockIdx.x * 256 + threadIdx.x;
  if (i < n) p[i] = make_uint4(0u, 0u, 0u, 0u);
}

// ---------------- graph prep + zero accumulator ----------------
__global__ void k_prep(const float* __restrict__ jg, const float* __restrict__ gs,
                       const float* __restrict__ gt, const float* __restrict__ sjcw,
                       const float* __restrict__ tjcw,
                       float* g_sg, float* g_mws, float* g_mwt,
                       float* g_rs, float* g_rt, float* acc) {
  int tid = threadIdx.x;
  for (int i = tid; i < H_*J_*J_; i += 256) {
    int h = i / (J_*J_);
    int j = i % J_;
    float g = jg[i];
    float s = fabsf(gs[i] * g);
    float t = fabsf(gt[i] * g);
    g_sg[i]  = s;                       // sg[h,k,j]
    g_mws[i] = s * sjcw[h*J_ + j];      // mw_s[h,i,j] = sg * s_jcw[h,j]
    g_mwt[i] = t * tjcw[h*J_ + j];
  }
  for (int i = tid; i < B_*H1_; i += 256) acc[i] = 0.f;
  __syncthreads();
  for (int i = tid; i < H_*J_; i += 256) {     // i = h*17 + row
    float rs = 0.f, rt = 0.f;
    for (int j = 0; j < J_; ++j) { rs += g_mws[i*J_ + j]; rt += g_mwt[i*J_ + j]; }
    g_rs[i] = rs; g_rt[i] = rt;
  }
}

// ---- stage A: per (b,t,d) build all f16 GEMM inputs (comm, d0, d1, copies) ----
__global__ void __launch_bounds__(256)
k_stageA(const float* __restrict__ fw, const float* __restrict__ fp,
         const float* __restrict__ g_sg, const float* __restrict__ g_mws,
         const float* __restrict__ g_mwt, const float* __restrict__ g_rs,
         const float* __restrict__ g_rt,
         _Float16* Xw, _Float16* Xdiff, _Float16* Xc0,
         _Float16* Xcomm, _Float16* Xd0, _Float16* Xd1) {
  __shared__ float s_sg[H_*J_*J_], s_mws[H_*J_*J_], s_mwt[H_*J_*J_];
  __shared__ float s_rs[H_*J_], s_rt[H_*J_];
  int tid = threadIdx.x;
  for (int i = tid; i < H_*J_*J_; i += 256) { s_sg[i]=g_sg[i]; s_mws[i]=g_mws[i]; s_mwt[i]=g_mwt[i]; }
  for (int i = tid; i < H_*J_;    i += 256) { s_rs[i]=g_rs[i]; s_rt[i]=g_rt[i]; }
  __syncthreads();
  int g = blockIdx.x * 256 + tid;
  if (g >= B_*T_*D_) return;
  int d = g % D_;
  int t = (g / D_) % T_;
  int b = g / (D_ * T_);
  int cd = d + ((d >= HALF_) ? (KP_ - HALF_) : 0);   // K-padded column
  int t2 = (t + 1 < T_) ? t + 1 : t;                 // temporal shift (edge clamp)
  float v[J_], vs[J_];
  #pragma unroll
  for (int j = 0; j < J_; ++j) {
    v[j]  = fp[((size_t)(b*J_ + j)*T_ + t )*D_ + d];
    vs[j] = fp[((size_t)(b*J_ + j)*T_ + t2)*D_ + d];
  }
  {
    float w0 = fw[(size_t)(b*T_ + t )*D_ + d];
    float w1 = fw[(size_t)(b*T_ + t2)*D_ + d];
    Xw   [(size_t)(b*T_+t)*RS_ + cd] = (_Float16)w0;
    Xdiff[(size_t)(b*T_+t)*RS_ + cd] = (_Float16)fabsf(w1 - w0);
  }
  #pragma unroll
  for (int j = 0; j < J_; ++j)
    Xc0[((size_t)(b*T_+t)*J_ + j)*RS_ + cd] = (_Float16)v[j];
  for (int h = 0; h < H_; ++h) {
    size_t rbase = ((size_t)(h*B_ + b)*T_ + t)*J_;
    for (int j = 0; j < J_; ++j) {
      float c = 0.f, e0 = 0.f, e1 = 0.f;
      const float* sgc  = &s_sg [(h*J_     )*J_ + j];  // sg[h,k,j], stride J_
      const float* mwr0 = &s_mws[(h*J_ + j )*J_    ];  // mw_s[h,j,:]
      const float* mwr1 = &s_mwt[(h*J_ + j )*J_    ];
      #pragma unroll
      for (int k = 0; k < J_; ++k) {
        c  += v[k]  * sgc[k*J_];
        e0 += v[k]  * mwr0[k];
        e1 += vs[k] * mwr1[k];
      }
      e0 -= v[j] * s_rs[h*J_ + j];
      e1 -= v[j] * s_rt[h*J_ + j];
      Xcomm[(rbase + j)*RS_ + cd] = (_Float16)c;
      Xd0  [(rbase + j)*RS_ + cd] = (_Float16)e0;
      Xd1  [(rbase + j)*RS_ + cd] = (_Float16)e1;
    }
  }
}

// ---- fused WMMA GEMM + bias + ReLU + weighted batch-sum (atomic into acc) ----
// X rows are K-padded (stride RS_, zeros in pad cols / pad rows), so all A
// loads are unconditional 16B-aligned b128 loads: no exec-mask divergence.
__global__ void __launch_bounds__(256)
k_gemm(const _Float16* __restrict__ X, int R, int rpb,
       const float* __restrict__ W, const float* __restrict__ bias,
       int colOff, float scale, float* __restrict__ acc) {
  __shared__ __attribute__((aligned(64))) _Float16 sW[H1_*KP_];  // f16 weights, K-padded
  __shared__ float sB[H1_];
  int tid = threadIdx.x;
  for (int i = tid; i < H1_*KP_; i += 256) {
    int n = i / KP_, k = i % KP_;
    sW[i] = (k < HALF_) ? (_Float16)W[n*HALF_ + k] : (_Float16)0.f;
  }
  if (tid < H1_) sB[tid] = bias[tid];
  __syncthreads();

  int lane = tid & 31, wave = tid >> 5;
  int Mbase = (blockIdx.x * 8 + wave) * 16;      // always < Rpad (grid sized to it)

  int m  = lane & 15;
  int hi = (lane >> 4) & 1;                      // half-wave selector
  const _Float16* xrow = X + (size_t)(Mbase + m) * RS_ + colOff + hi*8;

  // Preload all 7 K-tiles of A: lane layout (16-bit 16x32):
  // lanes<16 hold K 0-7 & 16-23; lanes>=16 hold K 8-15 & 24-31.
  union AV { v16h v; v8h h[2]; } a[7];
  #pragma unroll
  for (int kt = 0; kt < 7; ++kt) {
    a[kt].h[0] = *(const v8h*)(xrow + kt*32);
    a[kt].h[1] = *(const v8h*)(xrow + kt*32 + 16);
  }

  v8f c[8];
  #pragma unroll
  for (int nt = 0; nt < 8; ++nt)
    #pragma unroll
    for (int r = 0; r < 8; ++r) c[nt][r] = 0.f;

  const _Float16* swl = &sW[m*KP_ + hi*8];       // B: lane=N, K 0-7 & 16-23 / 8-15 & 24-31
  #pragma unroll
  for (int kt = 0; kt < 7; ++kt) {
    #pragma unroll
    for (int nt = 0; nt < 8; ++nt) {
      union AV bu; bu.h[0] = *(const v8h*)(swl + nt*16*KP_ + kt*32); bu.h[1] = *(const v8h*)(swl + nt*16*KP_ + kt*32 + 16);
      const v16h bvec = bu.v;
      c[nt] = __builtin_amdgcn_wmma_f32_16x16x32_f16(
          false, a[kt].v, false, bvec, (short)0, c[nt], false, false);
    }
  }

  // epilogue: C rows = Mbase + r + hi*8; batch b = (row/rpb)%16
  int rowsBase = Mbase + hi*8;
  int q   = rowsBase / rpb;
  int b0  = q & 15, b1 = (q + 1) & 15;
  int rem = (q + 1) * rpb - rowsBase;            // rows before b changes (>=1)
  #pragma unroll
  for (int nt = 0; nt < 8; ++nt) {
    int n = nt*16 + m;
    float bn = sB[n];
    float runA = 0.f, runB = 0.f;
    #pragma unroll
    for (int r = 0; r < 8; ++r) {
      int rw = rowsBase + r;
      float val = c[nt][r] + bn;
      val = val > 0.f ? val : 0.f;
      val = (rw < R) ? val : 0.f;                // mask padded rows (bias!)
      if (r < rem) runA += val; else runB += val;
    }
    __hip_atomic_fetch_add(&acc[b0*H1_ + n], runA * scale,
                           __ATOMIC_RELAXED, __HIP_MEMORY_SCOPE_AGENT);
    if (rem < 8)
      __hip_atomic_fetch_add(&acc[b1*H1_ + n], runB * scale,
                             __ATOMIC_RELAXED, __HIP_MEMORY_SCOPE_AGENT);
  }
}

// ---- final: out = relu( (acc/3060) @ (Wr[:, :128]+Wr[:, 128:]).T + b_reg ) ----
__global__ void k_final(const float* __restrict__ acc, const float* __restrict__ Wr,
                        const float* __restrict__ br, float* __restrict__ out) {
  int g = blockIdx.x * 256 + threadIdx.x;
  if (g >= B_ * 512) return;
  int b = g >> 9, o = g & 511;
  const float inv = 1.0f / 3060.0f;              // mean over T*15*J
  float s = br[o];
  const float* ar = acc + b*H1_;
  const float* wr = Wr + (size_t)o*256;
  for (int ci = 0; ci < H1_; ++ci)
    s += ar[ci] * inv * (wr[ci] + wr[ci + H1_]);
  out[(size_t)b*512 + o] = s > 0.f ? s : 0.f;
}

extern "C" void kernel_launch(void* const* d_in, const int* in_sizes, int n_in,
                              void* d_out, int out_size, void* d_ws, size_t ws_size,
                              hipStream_t stream) {
  (void)in_sizes; (void)n_in; (void)out_size; (void)ws_size;
  const float* fw      = (const float*)d_in[0];
  const float* fp      = (const float*)d_in[1];
  const float* jg      = (const float*)d_in[2];
  const float* gs      = (const float*)d_in[3];
  const float* gt      = (const float*)d_in[4];
  const float* sjcw    = (const float*)d_in[5];
  const float* tjcw    = (const float*)d_in[6];
  const float* W_whole = (const float*)d_in[7];
  const float* b_whole = (const float*)d_in[8];
  const float* W_diffw = (const float*)d_in[9];
  const float* b_diffw = (const float*)d_in[10];
  const float* W_comm0 = (const float*)d_in[11];
  const float* b_comm0 = (const float*)d_in[12];
  const float* W_diff0 = (const float*)d_in[13];
  const float* b_diff0 = (const float*)d_in[14];
  const float* W_diff1 = (const float*)d_in[15];
  const float* b_diff1 = (const float*)d_in[16];
  const float* W_reg   = (const float*)d_in[17];
  const float* b_reg   = (const float*)d_in[18];

  float* wsf   = (float*)d_ws;
  float* g_sg  = wsf;            // 1156
  float* g_mws = wsf + 1156;     // 1156
  float* g_mwt = wsf + 2312;     // 1156
  float* g_rs  = wsf + 3468;     // 68
  float* g_rt  = wsf + 3536;     // 68
  float* acc   = wsf + 4096;     // 16*128
  _Float16* h16   = (_Float16*)((char*)d_ws + 32768);
  _Float16* Xw    = h16;                                   // RW_PAD_   * RS_
  _Float16* Xdiff = Xw    + (size_t)RW_PAD_  * RS_;
  _Float16* Xc0   = Xdiff + (size_t)RW_PAD_  * RS_;        // RC0_PAD_  * RS_
  _Float16* Xcomm = Xc0   + (size_t)RC0_PAD_ * RS_;        // RBIG_PAD_ * RS_
  _Float16* Xd0   = Xcomm + (size_t)RBIG_PAD_* RS_;
  _Float16* Xd1   = Xd0   + (size_t)RBIG_PAD_* RS_;
  const size_t totalHalfs =
      (size_t)(2*RW_PAD_ + RC0_PAD_ + 3*RBIG_PAD_) * RS_;  // 19,267,584 (~36.8 MB)
  const int nVec16 = (int)(totalHalfs * sizeof(_Float16) / 16);

  k_zero<<<(nVec16 + 255)/256, 256, 0, stream>>>((uint4*)h16, nVec16);
  k_prep<<<1, 256, 0, stream>>>(jg, gs, gt, sjcw, tjcw, g_sg, g_mws, g_mwt, g_rs, g_rt, acc);
  k_stageA<<<300, 256, 0, stream>>>(fw, fp, g_sg, g_mws, g_mwt, g_rs, g_rt,
                                    Xw, Xdiff, Xc0, Xcomm, Xd0, Xd1);
  for (int idx = 0; idx < 2; ++idx) {
    int co = idx * KP_;
    const float* Ww = W_whole + idx*H1_*HALF_;  const float* bw = b_whole + idx*H1_;
    const float* Wd = W_diffw + idx*H1_*HALF_;  const float* bd = b_diffw + idx*H1_;
    const float* Wc = W_comm0 + idx*H1_*HALF_;  const float* bc = b_comm0 + idx*H1_;
    const float* W0 = W_diff0 + idx*H1_*HALF_;  const float* b0 = b_diff0 + idx*H1_;
    const float* W1 = W_diff1 + idx*H1_*HALF_;  const float* b1 = b_diff1 + idx*H1_;
    k_gemm<<<RW_PAD_/128,   256, 0, stream>>>(Xw,    192,   12,  Ww, bw, co, 17.f, acc); // ew
    k_gemm<<<RW_PAD_/128,   256, 0, stream>>>(Xdiff, 192,   12,  Wd, bd, co, 17.f, acc); // ed
    k_gemm<<<RC0_PAD_/128,  256, 0, stream>>>(Xc0,   3264,  204, Wc, bc, co, 1.f,  acc); // ec0
    k_gemm<<<RBIG_PAD_/128, 256, 0, stream>>>(Xcomm, 13056, 204, Wc, bc, co, 1.f,  acc); // ec1
    k_gemm<<<RBIG_PAD_/128, 256, 0, stream>>>(Xd0,   13056, 204, W0, b0, co, 1.f,  acc); // e0
    k_gemm<<<RBIG_PAD_/128, 256, 0, stream>>>(Xd1,   13056, 204, W1, b1, co, 1.f,  acc); // e1
  }
  k_final<<<32, 256, 0, stream>>>(acc, W_reg, b_reg, (float*)d_out);
}
